// RadarTimeNet_20444044329077
// MI455X (gfx1250) — matrix operand order repaired and checked by reading, not compiled, not run on hardware
//
#include <hip/hip_runtime.h>
#include <hip/hip_bf16.h>
#include <cmath>

typedef __attribute__((ext_vector_type(16))) _Float16 v16h;
typedef __attribute__((ext_vector_type(8)))  _Float16 v8h;
typedef __attribute__((ext_vector_type(8)))  float    v8f;
typedef __attribute__((ext_vector_type(4)))  unsigned int v4u;
typedef __attribute__((ext_vector_type(8)))  int v8i;
typedef __attribute__((ext_vector_type(4)))  int v4i;

constexpr int NB  = 16;    // batch
constexpr int NRX = 4;
constexpr int ND  = 64;    // chirps / doppler bins
constexpr int NS  = 256;   // samples / range bins
constexpr int NSP = 258;   // padded columns (1 zero col each side)

#if defined(__has_builtin)
#if __has_builtin(__builtin_amdgcn_tensor_load_to_lds) && \
    __has_builtin(__builtin_amdgcn_s_wait_tensorcnt)
#define HAVE_TDM 1
#endif
#endif

#ifdef HAVE_TDM
// Build a Tensor-DMA D# (CDNA5 ISA) and issue a 3-D tile load.
// dims/strides in data_size (=2 byte) units. X=dim0 contiguous, Y uses st0,
// Z uses st1. All accesses in-bounds by construction.
__device__ inline void tdm_load_3d(unsigned lds_off, const void* gptr,
                                   unsigned td0, unsigned tl0,
                                   unsigned long long st0, unsigned td1,
                                   unsigned tl1, unsigned long long st1,
                                   unsigned td2, unsigned tl2) {
  unsigned long long ga = (unsigned long long)(size_t)gptr;
  v4u g0;
  g0[0] = 1u;                                   // count=1, user descriptor
  g0[1] = lds_off;                              // lds_addr (bytes)
  g0[2] = (unsigned)(ga & 0xffffffffull);       // global_addr[31:0]
  g0[3] = (unsigned)((ga >> 32) & 0x01ffffffull) | (2u << 30);  // type=2
  v8i g1;
  g1[0] = (int)(1u << 16);                      // data_size = 2 bytes
  g1[1] = (int)((td0 & 0xffffu) << 16);         // tensor_dim0[15:0]
  g1[2] = (int)(((td0 >> 16) & 0xffffu) | ((td1 & 0xffffu) << 16));
  g1[3] = (int)(((td1 >> 16) & 0xffffu) | ((tl0 & 0xffffu) << 16));
  g1[4] = (int)((tl1 & 0xffffu) | ((tl2 & 0xffffu) << 16));
  g1[5] = (int)(st0 & 0xffffffffull);           // dim0_stride[31:0]
  g1[6] = (int)(((st0 >> 32) & 0xffffull) | ((st1 & 0xffffull) << 16));
  g1[7] = (int)((st1 >> 16) & 0xffffffffull);   // dim1_stride[47:16]
  v4i g2;
  g2[0] = (int)td2;                             // tensor_dim2
  g2[1] = 1;                                    // tensor_dim3
  g2[2] = 0;                                    // dim2_stride (tile_dim3=0)
  g2[3] = 0;
  v4i g3 = {0, 0, 0, 0};
#if __clang_major__ >= 23
  v8i g4 = {0, 0, 0, 0, 0, 0, 0, 0};
  __builtin_amdgcn_tensor_load_to_lds(g0, g1, g2, g3, g4, 0);
#else
  __builtin_amdgcn_tensor_load_to_lds(g0, g1, g2, g3, 0);
#endif
}
#endif

__device__ inline v16h pack_ab(v8h lo, v8h hi) {
  v16h r;
#pragma unroll
  for (int t = 0; t < 8; ++t) { r[t] = lo[t]; r[t + 8] = hi[t]; }
  return r;
}

// ---------------------------------------------------------------------------
// 1-row conv(3) as WMMA GEMM, <=2 (input, weight) terms, double-buffered TDM.
// in*: f16 (NB, 258, Cin) sample-major, cols 0 & 257 zero (halo);
// w*:  f16 (Cout, 3, Cin) kx-major; out: f16 (NB, 258, Cout), writes cols 1..256.
// Block (32,4): waves share M-tile row0; wave w covers cols [w*64, w*64+64)
// with 4 accumulators. A tile (16x3x32) and B tile (258x32) staged in LDS by
// the Tensor Data Mover, pipelined one stage ahead. Grid (Cout/16, 1, NB).
// ---------------------------------------------------------------------------
__global__ __launch_bounds__(128) void conv1d_wmma_kernel(
    const _Float16* __restrict__ in0, const _Float16* __restrict__ w0,
    const _Float16* __restrict__ in1, const _Float16* __restrict__ w1,
    int nterms, int Cin, int Cout, const float* __restrict__ bias,
    const float* __restrict__ bns, const float* __restrict__ bnt,
    _Float16* __restrict__ out) {
  __shared__ _Float16 Alds[2][16 * 3 * 32];
  __shared__ _Float16 Blds[2][NSP * 32];
  const int lane = threadIdx.x, wv = threadIdx.y;
  const int row0 = blockIdx.x * 16;
  const int b    = blockIdx.z;
  const int n0w  = wv * 64;
  const int r    = lane & 15;
  const int kbA  = (lane < 16) ? 0 : 1;   // A chunk select (16 halves)
  const int cidx = lane & 15;
  const int kbB  = (lane < 16) ? 0 : 16;  // B k-offset
  const int spt  = Cin >> 5;              // stages per term
  const int nstages = nterms * spt;

  auto load_stage = [&](int stage, int sel) {
    int term = stage / spt;
    int i0   = (stage - term * spt) * 32;
    const _Float16* W  = (term == 0) ? w0 : w1;
    const _Float16* IN = ((term == 0) ? in0 : in1) + (size_t)b * NSP * Cin;
#ifdef HAVE_TDM
    if (wv == 0) {
      // A: X=k(32 of Cin), Y=kx(3, stride Cin), Z=row(16, stride 3*Cin)
      tdm_load_3d((unsigned)(size_t)&Alds[sel][0],
                  W + ((size_t)row0 * 3) * Cin + i0, (unsigned)Cin, 32u,
                  (unsigned long long)Cin, 3u, 3u,
                  (unsigned long long)(3 * Cin), 16u, 16u);
      // B: X=k(32 of Cin), Y=col(258, stride Cin)
      tdm_load_3d((unsigned)(size_t)&Blds[sel][0], IN + i0, (unsigned)Cin, 32u,
                  (unsigned long long)Cin, (unsigned)NSP, (unsigned)NSP, 0ull,
                  1u, 0u);
    }
#else
    int tid = wv * 32 + lane;
    for (int e = tid; e < 192; e += 128) {
      int r_ = e / 12, rem = e % 12, kx_ = rem >> 2, c_ = rem & 3;
      ((v8h*)&Alds[sel][0])[e] = *(const v8h*)(
          W + ((size_t)(row0 + r_) * 3 + kx_) * Cin + i0 + c_ * 8);
    }
    for (int e = tid; e < NSP * 4; e += 128) {
      int col_ = e >> 2, c_ = e & 3;
      ((v8h*)&Blds[sel][0])[e] =
          *(const v8h*)(IN + (size_t)col_ * Cin + i0 + c_ * 8);
    }
#endif
  };

  v8f acc[4] = {};
  load_stage(0, 0);
#ifdef HAVE_TDM
  if (wv == 0) __builtin_amdgcn_s_wait_tensorcnt((short)0);
#endif
  __syncthreads();
  for (int stage = 0; stage < nstages; ++stage) {
    int sel = stage & 1;
    if (stage + 1 < nstages) load_stage(stage + 1, sel ^ 1);  // async prefetch
#pragma unroll
    for (int kx = 0; kx < 3; ++kx) {
      const v8h* ap = (const v8h*)(&Alds[sel][(r * 3 + kx) * 32]);
      v16h a = pack_ab(ap[2 * kbA], ap[2 * kbA + 1]);
#pragma unroll
      for (int j = 0; j < 4; ++j) {
        int coll = n0w + j * 16 + cidx + kx;  // 0..257, always in-bounds
        const v8h* p = (const v8h*)(&Blds[sel][coll * 32 + kbB]);
        v16h bb = pack_ab(p[0], p[1]);
        acc[j] = __builtin_amdgcn_wmma_f32_16x16x32_f16(
            false, a, false, bb, (short)0, acc[j], false, false);
      }
    }
#ifdef HAVE_TDM
    if (wv == 0 && stage + 1 < nstages)
      __builtin_amdgcn_s_wait_tensorcnt((short)0);
#endif
    __syncthreads();
  }
  const int mbase = row0 + ((lane < 16) ? 0 : 8);
  float bs[8], sc[8], tt[8];
#pragma unroll
  for (int v = 0; v < 8; ++v) {
    bs[v] = bias[mbase + v]; sc[v] = bns[mbase + v]; tt[v] = bnt[mbase + v];
  }
#pragma unroll
  for (int j = 0; j < 4; ++j) {
    v8h ov;
#pragma unroll
    for (int v = 0; v < 8; ++v) {
      float val = acc[j][v] + bs[v];
      val = val * sc[v] + tt[v];
      val = val > 0.0f ? val : 0.0f;
      ov[v] = (_Float16)val;
    }
    int col = n0w + j * 16 + cidx;
    *(v8h*)(out + ((size_t)b * NSP + col + 1) * Cout + mbase) = ov;
  }
}

// ---------------------------------------------------------------------------
// Plain WMMA GEMM: C(MxN,f32) = A(MxK,f16 row-major) @ Bt(NxK,f16)^T
// Block (32,4), waves share M-tile (A staged via TDM); grid (M/16, 1, 1).
// ---------------------------------------------------------------------------
__global__ __launch_bounds__(128) void gemm_wmma_kernel(
    const _Float16* __restrict__ A, const _Float16* __restrict__ Bt,
    float* __restrict__ C, int N, int K) {
  __shared__ _Float16 Alds[16 * 32];
  const int lane = threadIdx.x, wv = threadIdx.y;
  const int row0 = blockIdx.x * 16;
  const int n0w  = wv * 64;
  const int r    = lane & 15;
  const int kbA  = (lane < 16) ? 0 : 1;
  const int cidx = lane & 15;
  const int kbB  = (lane < 16) ? 0 : 16;
  v8f acc[4] = {};
  for (int k0 = 0; k0 < K; k0 += 32) {
    __syncthreads();
#ifdef HAVE_TDM
    if (wv == 0) {
      tdm_load_3d((unsigned)(size_t)&Alds[0], A + (size_t)row0 * K + k0,
                  (unsigned)K, 32u, (unsigned long long)K, 16u, 16u, 0ull, 1u,
                  0u);
      __builtin_amdgcn_s_wait_tensorcnt((short)0);
    }
#else
    {
      int tid = wv * 32 + lane;
      for (int e = tid; e < 64; e += 128) {
        int r_ = e >> 2, c_ = e & 3;
        ((v8h*)Alds)[e] =
            *(const v8h*)(A + (size_t)(row0 + r_) * K + k0 + c_ * 8);
      }
    }
#endif
    __syncthreads();
    const v8h* ap = (const v8h*)(Alds + r * 32);
    v16h a = pack_ab(ap[2 * kbA], ap[2 * kbA + 1]);
#pragma unroll
    for (int j = 0; j < 4; ++j) {
      int col = n0w + j * 16 + cidx;
      const v8h* p = (const v8h*)(Bt + (size_t)col * K + k0 + kbB);
      v16h bb = pack_ab(p[0], p[1]);
      acc[j] = __builtin_amdgcn_wmma_f32_16x16x32_f16(false, a, false, bb,
                                                      (short)0, acc[j], false,
                                                      false);
    }
  }
#pragma unroll
  for (int j = 0; j < 4; ++j) {
    int col = n0w + j * 16 + cidx;
#pragma unroll
    for (int v = 0; v < 8; ++v) {
      int m = row0 + ((lane < 16) ? v : 8 + v);
      C[(size_t)m * N + col] = acc[j][v];
    }
  }
}

// ---------------------------------------------------------------------------
// Fused tc1 -> tc2 -> rx front-end; writes f16 (B, 258, 2048) sample-major
// (columns shifted +1 for halo) with channel index co*64 + d.
// ---------------------------------------------------------------------------
__global__ __launch_bounds__(256) void frontend_kernel(
    const float* __restrict__ x, const float* __restrict__ tc1w,
    const float* __restrict__ tc1b, const float* __restrict__ tc1s,
    const float* __restrict__ tc1t, const float* __restrict__ tc2w,
    const float* __restrict__ tc2b, const float* __restrict__ tc2s,
    const float* __restrict__ tc2t, const float* __restrict__ rxw,
    const float* __restrict__ rxb, const float* __restrict__ rxs,
    const float* __restrict__ rxt, _Float16* __restrict__ out) {
  __shared__ float sw1[96], sb1[16], ss1[16], st1[16];
  __shared__ float sw2[1536], sb2[32], ss2[32], st2[32];
  __shared__ float swr[4096], sbr[32], ssr[32], str_[32];
  int tid = threadIdx.x, nt = blockDim.x;
  for (int i = tid; i < 96; i += nt) sw1[i] = tc1w[i];
  for (int i = tid; i < 16; i += nt) { sb1[i] = tc1b[i]; ss1[i] = tc1s[i]; st1[i] = tc1t[i]; }
  for (int i = tid; i < 1536; i += nt) sw2[i] = tc2w[i];
  for (int i = tid; i < 32; i += nt) {
    sb2[i] = tc2b[i]; ss2[i] = tc2s[i]; st2[i] = tc2t[i];
    sbr[i] = rxb[i];  ssr[i] = rxs[i];  str_[i] = rxt[i];
  }
  for (int i = tid; i < 4096; i += nt) swr[i] = rxw[i];
  __syncthreads();

  int gid = blockIdx.x * blockDim.x + threadIdx.x;
  int s = gid & 255, d = (gid >> 8) & 63, b = gid >> 14;

  float acc[32];
#pragma unroll
  for (int c = 0; c < 32; ++c) acc[c] = sbr[c];

  for (int rxi = 0; rxi < NRX; ++rxi) {
    const float* xb = x + ((size_t)((b * NRX + rxi) * ND + d)) * NS * 2;
    float t1[3][16];
#pragma unroll
    for (int kk = 0; kk < 3; ++kk) {
      int sc = s + kk - 1;
      if (sc < 0 || sc >= NS) {
#pragma unroll
        for (int c = 0; c < 16; ++c) t1[kk][c] = 0.0f;
        continue;
      }
      float xv[3][2];
#pragma unroll
      for (int kt = 0; kt < 3; ++kt) {
        int si = sc + kt - 1;
        if (si < 0 || si >= NS) { xv[kt][0] = 0.0f; xv[kt][1] = 0.0f; }
        else { xv[kt][0] = xb[si * 2]; xv[kt][1] = xb[si * 2 + 1]; }
      }
#pragma unroll
      for (int c = 0; c < 16; ++c) {
        float a = sb1[c];
#pragma unroll
        for (int j = 0; j < 2; ++j)
#pragma unroll
          for (int kt = 0; kt < 3; ++kt)
            a += xv[kt][j] * sw1[(c * 2 + j) * 3 + kt];
        a = a * ss1[c] + st1[c];
        t1[kk][c] = a > 0.0f ? a : 0.0f;
      }
    }
    for (int c2 = 0; c2 < 32; ++c2) {
      float a = sb2[c2];
#pragma unroll
      for (int c1 = 0; c1 < 16; ++c1)
#pragma unroll
        for (int kk = 0; kk < 3; ++kk)
          a += t1[kk][c1] * sw2[(c2 * 16 + c1) * 3 + kk];
      a = a * ss2[c2] + st2[c2];
      a = a > 0.0f ? a : 0.0f;
#pragma unroll
      for (int co = 0; co < 32; ++co)
        acc[co] += a * swr[(co * 32 + c2) * 4 + rxi];
    }
  }
#pragma unroll
  for (int co = 0; co < 32; ++co) {
    float a = acc[co] * ssr[co] + str_[co];
    a = a > 0.0f ? a : 0.0f;
    out[((size_t)b * NSP + s + 1) * 2048 + co * ND + d] = (_Float16)a;
  }
}

// zero the two halo columns (0 and 257) of a padded (NB, 258, C) f16 buffer
__global__ void zero_pad_kernel(_Float16* __restrict__ buf, int C) {
  int idx = blockIdx.x * blockDim.x + threadIdx.x;
  if (idx >= NB * 2 * C) return;
  int c = idx % C, rem = idx / C;
  int side = rem & 1, b = rem >> 1;
  buf[((size_t)b * NSP + (side ? NSP - 1 : 0)) * C + c] = (_Float16)0.0f;
}

// weff[(o*3+kx)*Cin + i] = f16( sum_{ky in mask} w[((o*Cin+i)*3+ky)*3+kx] )
__global__ void combine_w_kernel(const float* __restrict__ w,
                                 _Float16* __restrict__ dst, int Cout, int Cin,
                                 int mask) {
  int idx = blockIdx.x * blockDim.x + threadIdx.x;
  if (idx >= Cout * 3 * Cin) return;
  int i = idx % Cin, rem = idx / Cin;
  int kx = rem % 3, o = rem / 3;
  float a = 0.0f;
#pragma unroll
  for (int ky = 0; ky < 3; ++ky)
    if ((mask >> ky) & 1) a += w[((o * Cin + i) * 3 + ky) * 3 + kx];
  dst[idx] = (_Float16)a;
}

// wr{r,i}T[r*256+s] = f16(wr[s,r,{0,1}]); grouped doppler weights wdc (5,64,2)
__global__ void prep_fft_kernel(const float* __restrict__ wr,
                                const float* __restrict__ wd,
                                _Float16* __restrict__ wrrT,
                                _Float16* __restrict__ wriT,
                                float* __restrict__ wdc) {
  int idx = blockIdx.x * blockDim.x + threadIdx.x;
  if (idx < NS * NS) {
    int rr = idx / NS, s = idx % NS;
    wrrT[idx] = (_Float16)wr[((size_t)s * NS + rr) * 2 + 0];
    wriT[idx] = (_Float16)wr[((size_t)s * NS + rr) * 2 + 1];
  }
  if (idx < ND * 2) {
    int comp = idx & 1, d = idx >> 1;
    wdc[(0 * ND + d) * 2 + comp] = wd[(0 * ND + d) * 2 + comp];
    wdc[(1 * ND + d) * 2 + comp] = wd[(1 * ND + d) * 2 + comp];
    float s = 0.0f;
    for (int c = 2; c <= 61; ++c) s += wd[(c * ND + d) * 2 + comp];
    wdc[(2 * ND + d) * 2 + comp] = s;
    wdc[(3 * ND + d) * 2 + comp] = wd[(62 * ND + d) * 2 + comp];
    wdc[(4 * ND + d) * 2 + comp] = wd[(63 * ND + d) * 2 + comp];
  }
}

// 1x1 conv + sigmoid on the 5 distinct up2 row variants -> xr5 f32 + f16 copy
__global__ void up3_sigmoid_kernel(const _Float16* __restrict__ up2v,
                                   const float* __restrict__ w,
                                   const float* __restrict__ bptr,
                                   float* __restrict__ xr5,
                                   _Float16* __restrict__ xr5h) {
  int idx = blockIdx.x * blockDim.x + threadIdx.x;
  if (idx >= NB * 5 * NS) return;
  int s = idx & 255, r8 = idx >> 8;
  int v = r8 % 5, b = r8 / 5;
  const _Float16* src = up2v + (((size_t)v * NB + b) * NSP + s + 1) * 32;
  float acc = bptr[0];
#pragma unroll
  for (int i = 0; i < 32; ++i) acc += (float)src[i] * w[i];
  float sg = 1.0f / (1.0f + __expf(-acc));
  xr5[((size_t)b * 5 + v) * NS + s]  = sg;
  xr5h[((size_t)b * 5 + v) * NS + s] = (_Float16)sg;
}

// doppler combine (5 groups), magnitude, blend + clip, NHWC output
__global__ void final_kernel(const float* __restrict__ rr5,
                             const float* __restrict__ ri5,
                             const float* __restrict__ wdc,
                             const float* __restrict__ xr5,
                             float* __restrict__ out) {
  int idx = blockIdx.x * blockDim.x + threadIdx.x;
  if (idx >= NB * ND * NS) return;
  int s = idx & 255, d = (idx >> 8) & 63, b = idx >> 14;
  int v = (d == 0) ? 0 : (d == 1) ? 1 : (d <= 61) ? 2 : (d == 62) ? 3 : 4;
  float rdr = 0.0f, rdi = 0.0f;
#pragma unroll
  for (int g = 0; g < 5; ++g) {
    float rr = rr5[((size_t)b * 5 + g) * NS + s];
    float ri = ri5[((size_t)b * 5 + g) * NS + s];
    float w0 = wdc[(g * ND + d) * 2 + 0];
    float w1 = wdc[(g * ND + d) * 2 + 1];
    rdr += rr * w0 - ri * w1;
    rdi += rr * w1 + ri * w0;
  }
  float mag = sqrtf(rdr * rdr + rdi * rdi) * (1.0f / 16384.0f);
  float h = xr5[((size_t)b * 5 + v) * NS + s];
  float o = h + 0.1f * mag;
  o = fminf(fmaxf(o, 0.0f), 1.0f);
  out[((size_t)b * ND + d) * NS + s] = o;
}

// ---------------------------------------------------------------------------
extern "C" void kernel_launch(void* const* d_in, const int* in_sizes, int n_in,
                              void* d_out, int out_size, void* d_ws,
                              size_t ws_size, hipStream_t stream) {
  (void)in_sizes; (void)n_in; (void)out_size; (void)ws_size;
  const float* x    = (const float*)d_in[0];
  const float* tc1w = (const float*)d_in[1];
  const float* tc1b = (const float*)d_in[2];
  const float* tc1s = (const float*)d_in[3];
  const float* tc1t = (const float*)d_in[4];
  const float* tc2w = (const float*)d_in[5];
  const float* tc2b = (const float*)d_in[6];
  const float* tc2s = (const float*)d_in[7];
  const float* tc2t = (const float*)d_in[8];
  const float* rxw  = (const float*)d_in[9];
  const float* rxb  = (const float*)d_in[10];
  const float* rxs  = (const float*)d_in[11];
  const float* rxt  = (const float*)d_in[12];
  const float* cc1w = (const float*)d_in[13];
  const float* cc1b = (const float*)d_in[14];
  const float* cc1s = (const float*)d_in[15];
  const float* cc1t = (const float*)d_in[16];
  const float* cc2w = (const float*)d_in[17];
  const float* cc2b = (const float*)d_in[18];
  const float* cc2s = (const float*)d_in[19];
  const float* cc2t = (const float*)d_in[20];
  const float* rd1w = (const float*)d_in[21];
  const float* rd1b = (const float*)d_in[22];
  const float* rd1s = (const float*)d_in[23];
  const float* rd1t = (const float*)d_in[24];
  const float* rd2w = (const float*)d_in[25];
  const float* rd2b = (const float*)d_in[26];
  const float* rd2s = (const float*)d_in[27];
  const float* rd2t = (const float*)d_in[28];
  const float* up1w = (const float*)d_in[29];
  const float* up1b = (const float*)d_in[30];
  const float* up1s = (const float*)d_in[31];
  const float* up1t = (const float*)d_in[32];
  const float* up2w = (const float*)d_in[33];
  const float* up2b = (const float*)d_in[34];
  const float* up2s = (const float*)d_in[35];
  const float* up2t = (const float*)d_in[36];
  const float* up3w = (const float*)d_in[37];
  const float* up3b = (const float*)d_in[38];
  const float* wr   = (const float*)d_in[39];
  const float* wd   = (const float*)d_in[40];

  char* wsb = (char*)d_ws;
  size_t off = 0;
  auto carve = [&](size_t bytes) -> void* {
    void* p = wsb + off;
    off += (bytes + 255) & ~(size_t)255;
    return p;
  };
  _Float16* buf3   = (_Float16*)carve((size_t)NB * NSP * 2048 * 2);
  _Float16* wf_cc1 = (_Float16*)carve((size_t)64 * 3 * 2048 * 2);
  _Float16* wf_cc2 = (_Float16*)carve((size_t)64 * 3 * 64 * 2);
  _Float16* wf_rd1 = (_Float16*)carve((size_t)128 * 3 * 64 * 2);
  _Float16* wf_rd2 = (_Float16*)carve((size_t)128 * 3 * 128 * 2);
  _Float16* wf_up1 = (_Float16*)carve((size_t)3 * 64 * 3 * 128 * 2);
  _Float16* wf_up2 = (_Float16*)carve((size_t)9 * 32 * 3 * 64 * 2);
  _Float16* cc1o   = (_Float16*)carve((size_t)NB * NSP * 64 * 2);
  _Float16* cc2o   = (_Float16*)carve((size_t)NB * NSP * 64 * 2);
  _Float16* rd1o   = (_Float16*)carve((size_t)NB * NSP * 128 * 2);
  _Float16* rd2o   = (_Float16*)carve((size_t)NB * NSP * 128 * 2);
  _Float16* up1v   = (_Float16*)carve((size_t)3 * NB * NSP * 64 * 2);
  _Float16* up2v   = (_Float16*)carve((size_t)5 * NB * NSP * 32 * 2);
  float*    xr5    = (float*)carve((size_t)NB * 5 * NS * 4);
  _Float16* xr5h   = (_Float16*)carve((size_t)NB * 5 * NS * 2);
  float*    rr5    = (float*)carve((size_t)NB * 5 * NS * 4);
  float*    ri5    = (float*)carve((size_t)NB * 5 * NS * 4);
  _Float16* wrrT   = (_Float16*)carve((size_t)NS * NS * 2);
  _Float16* wriT   = (_Float16*)carve((size_t)NS * NS * 2);
  float*    wdc    = (float*)carve((size_t)5 * ND * 2 * 4);

  auto zpad = [&](_Float16* bufp, int C) {
    int n = NB * 2 * C;
    zero_pad_kernel<<<(n + 255) / 256, 256, 0, stream>>>(bufp, C);
  };
  zpad(buf3, 2048);
  zpad(cc1o, 64); zpad(cc2o, 64); zpad(rd1o, 128); zpad(rd2o, 128);
  for (int t = 0; t < 3; ++t) zpad(up1v + (size_t)t * NB * NSP * 64, 64);
  for (int v = 0; v < 5; ++v) zpad(up2v + (size_t)v * NB * NSP * 32, 32);

  auto comb = [&](const float* w, _Float16* dst, int Cout_, int Cin_, int m) {
    int n = Cout_ * 3 * Cin_;
    combine_w_kernel<<<(n + 255) / 256, 256, 0, stream>>>(w, dst, Cout_, Cin_, m);
  };
  comb(cc1w, wf_cc1, 64, 2048, 2);   // H=1 images: only middle ky
  comb(cc2w, wf_cc2, 64, 64, 2);
  comb(rd1w, wf_rd1, 128, 64, 2);
  comb(rd2w, wf_rd2, 128, 128, 2);
  const int up1_masks[3] = {6, 7, 3};  // top / mid / bottom rows
  for (int t = 0; t < 3; ++t)
    comb(up1w, wf_up1 + (size_t)t * 64 * 3 * 128, 64, 128, up1_masks[t]);
  const int up2_masks[9]  = {2, 4, 1, 6, 7, 3, 4, 1, 2};
  const int up2_inputs[9] = {0, 1, 0, 1, 1, 1, 2, 1, 2};
  for (int t = 0; t < 9; ++t)
    comb(up2w, wf_up2 + (size_t)t * 32 * 3 * 64, 32, 64, up2_masks[t]);

  prep_fft_kernel<<<(NS * NS + 255) / 256, 256, 0, stream>>>(wr, wd, wrrT,
                                                             wriT, wdc);

  frontend_kernel<<<(NB * ND * NS) / 256, 256, 0, stream>>>(
      x, tc1w, tc1b, tc1s, tc1t, tc2w, tc2b, tc2s, tc2t, rxw, rxb, rxs, rxt,
      buf3);

  dim3 blk(32, 4);
  conv1d_wmma_kernel<<<dim3(4, 1, NB), blk, 0, stream>>>(
      buf3, wf_cc1, nullptr, nullptr, 1, 2048, 64, cc1b, cc1s, cc1t, cc1o);
  conv1d_wmma_kernel<<<dim3(4, 1, NB), blk, 0, stream>>>(
      cc1o, wf_cc2, nullptr, nullptr, 1, 64, 64, cc2b, cc2s, cc2t, cc2o);
  conv1d_wmma_kernel<<<dim3(8, 1, NB), blk, 0, stream>>>(
      cc2o, wf_rd1, nullptr, nullptr, 1, 64, 128, rd1b, rd1s, rd1t, rd1o);
  conv1d_wmma_kernel<<<dim3(8, 1, NB), blk, 0, stream>>>(
      rd1o, wf_rd2, nullptr, nullptr, 1, 128, 128, rd2b, rd2s, rd2t, rd2o);

  for (int t = 0; t < 3; ++t)
    conv1d_wmma_kernel<<<dim3(4, 1, NB), blk, 0, stream>>>(
        rd2o, wf_up1 + (size_t)t * 64 * 3 * 128, nullptr, nullptr, 1, 128, 64,
        up1b, up1s, up1t, up1v + (size_t)t * NB * NSP * 64);

  const int termcnt[5] = {2, 2, 1, 2, 2};
  int slot = 0;
  for (int v = 0; v < 5; ++v) {
    const _Float16* i0p = up1v + (size_t)up2_inputs[slot] * NB * NSP * 64;
    const _Float16* w0p = wf_up2 + (size_t)slot * 32 * 3 * 64;
    const _Float16* i1p = nullptr;
    const _Float16* w1p = nullptr;
    if (termcnt[v] == 2) {
      i1p = up1v + (size_t)up2_inputs[slot + 1] * NB * NSP * 64;
      w1p = wf_up2 + (size_t)(slot + 1) * 32 * 3 * 64;
    }
    conv1d_wmma_kernel<<<dim3(2, 1, NB), blk, 0, stream>>>(
        i0p, w0p, i1p, w1p, termcnt[v], 64, 32, up2b, up2s, up2t,
        up2v + (size_t)v * NB * NSP * 32);
    slot += termcnt[v];
  }

  up3_sigmoid_kernel<<<(NB * 5 * NS + 255) / 256, 256, 0, stream>>>(
      up2v, up3w, up3b, xr5, xr5h);

  gemm_wmma_kernel<<<dim3(5, 1, 1), blk, 0, stream>>>(xr5h, wrrT, rr5, NS, NS);
  gemm_wmma_kernel<<<dim3(5, 1, 1), blk, 0, stream>>>(xr5h, wriT, ri5, NS, NS);

  final_kernel<<<(NB * ND * NS + 255) / 256, 256, 0, stream>>>(
      rr5, ri5, wdc, xr5, (float*)d_out);
}
